// EdgePredictor_56719338111193
// MI455X (gfx1250) — matrix operand order repaired and checked by reading, not compiled, not run on hardware
//
#include <hip/hip_runtime.h>
#include <hip/hip_bf16.h>

typedef __attribute__((ext_vector_type(16))) _Float16 v16h;
typedef __attribute__((ext_vector_type(8)))  float    v8f;
typedef __attribute__((ext_vector_type(2)))  float    v2f;

#define NV   4096
#define NE   24576
#define KNN  15
#define NEE  (NE + NV * KNN)   // 86016
#define HID  64
#define BIGF 3.402823e38f

// ---------------------------------------------------------------- utilities
__device__ inline void atomicMaxF(float* addr, float val) {
  unsigned int* ua = (unsigned int*)addr;
  unsigned int old = *ua;
  while (__uint_as_float(old) < val) {
    unsigned int assumed = old;
    old = atomicCAS(ua, assumed, __float_as_uint(val));
    if (old == assumed) break;
  }
}

__global__ void fill_f32_kernel(float* p, float v, int n) {
  int i = blockIdx.x * blockDim.x + threadIdx.x;
  if (i < n) p[i] = v;
}
__global__ void fill_i32_kernel(int* p, int v, int n) {
  int i = blockIdx.x * blockDim.x + threadIdx.x;
  if (i < n) p[i] = v;
}
__global__ void zero4_kernel(float4* __restrict__ p, int n4) {
  int i = blockIdx.x * blockDim.x + threadIdx.x;
  if (i < n4) p[i] = make_float4(0.f, 0.f, 0.f, 0.f);
}
__global__ void sq_kernel(const float* __restrict__ x, float* __restrict__ sq) {
  int i = blockIdx.x * blockDim.x + threadIdx.x;
  if (i < NV) {
    float a = x[i*3+0], b = x[i*3+1], c = x[i*3+2];
    sq[i] = a*a + b*b + c*c;
  }
}

// --------------------------------------------- stage 1a: D2 tiles via WMMA f32
// One block = 16-row tile; 8 waves each cover 512 columns (32 WMMAs).
// x (48 KB) is staged once into LDS so the inner loop is ds-fed, no divergent
// global gathers.
__global__ void dist_wmma_kernel(const float* __restrict__ x,
                                 const float* __restrict__ sq,
                                 float* __restrict__ D2) {
  __shared__ float sx[NV * 3];   // 48 KB
  for (int i = threadIdx.x; i < NV * 3; i += 256) sx[i] = x[i];
  __syncthreads();

  const int wave = threadIdx.x >> 5;
  const int lane = threadIdx.x & 31;
  const int hf   = lane >> 4;     // which 16-lane half
  const int l16  = lane & 15;
  const int r0   = blockIdx.x * 16;

  // A fragment: 16x4 f32, lane<16 holds (M=l16, K=0/1), lane>=16 holds (M=l16, K=2/3)
  const float rx0 = sx[(r0 + l16)*3 + 0];
  const float rx1 = sx[(r0 + l16)*3 + 1];
  const float rx2 = sx[(r0 + l16)*3 + 2];
  v2f a;
  a.x = hf ? rx2 : rx0;
  a.y = hf ? 0.0f : rx1;

  float sqr[8];
#pragma unroll
  for (int r = 0; r < 8; ++r) sqr[r] = sq[r0 + r + 8*hf];

  for (int ct = wave * 32; ct < wave * 32 + 32; ++ct) {
    const int c0 = ct * 16;
    // B fragment: 4x16 f32, v0 = rows K=0 (lanes0-15) / K=1 (lanes16-31), v1 = K=2 / K=3
    const float cx0 = sx[(c0 + l16)*3 + 0];
    const float cx1 = sx[(c0 + l16)*3 + 1];
    const float cx2 = sx[(c0 + l16)*3 + 2];
    v2f b;
    b.x = hf ? cx1 : cx0;
    b.y = hf ? 0.0f : cx2;
    v8f c = {};
    c = __builtin_amdgcn_wmma_f32_16x16x4_f32(false, a, false, b, (short)0, c, false, false);
    const float sqc = sq[c0 + l16];
#pragma unroll
    for (int r = 0; r < 8; ++r) {
      const int row  = r0 + r + 8*hf;
      const int coln = c0 + l16;
      float d2 = sqr[r] + sqc - 2.0f * c[r];
      if (row == coln) d2 = BIGF;
      D2[(size_t)row * NV + coln] = d2;
    }
  }
}

// --------------------------------------------- stage 1b: top-K select per row
__global__ void knn_select_kernel(const float* __restrict__ D2, int* __restrict__ knn) {
  __shared__ float sd[NV];
  __shared__ float rv[256];
  __shared__ int   ri[256];
  const int v = blockIdx.x;
  const int t = threadIdx.x;
  for (int j = t; j < NV; j += 256) sd[j] = D2[(size_t)v * NV + j];
  __syncthreads();
  for (int r = 0; r < KNN; ++r) {
    float best = BIGF; int bi = 0;
    for (int j = t; j < NV; j += 256) {
      float d = sd[j];
      if (d < best) { best = d; bi = j; }
    }
    rv[t] = best; ri[t] = bi;
    __syncthreads();
    for (int s = 128; s > 0; s >>= 1) {
      if (t < s && rv[t+s] < rv[t]) { rv[t] = rv[t+s]; ri[t] = ri[t+s]; }
      __syncthreads();
    }
    if (t == 0) { knn[v * KNN + r] = ri[0]; sd[ri[0]] = BIGF; }
    __syncthreads();
  }
}

// --------------------------------------------- stage 2: DevConv edge max
__global__ void devconv_kernel(const float* __restrict__ x,
                               const long long* __restrict__ erow,
                               const long long* __restrict__ ecol,
                               const int* __restrict__ knn,
                               const float* __restrict__ Wt,
                               const float* __restrict__ bt,
                               float* __restrict__ agg) {
  __shared__ float sW[HID * 3];
  __shared__ float sb[HID];
  for (int i = threadIdx.x; i < HID * 3; i += blockDim.x) sW[i] = Wt[i];
  if (threadIdx.x < HID) sb[threadIdx.x] = bt[threadIdx.x];
  __syncthreads();
  int e = blockIdx.x * blockDim.x + threadIdx.x;
  if (e >= NEE) return;
  int r, c;
  if (e < NE) { r = (int)erow[e]; c = (int)ecol[e]; }
  else        { int t2 = e - NE; r = t2 / KNN; c = knn[t2]; }
  float rx = x[r*3+0] - x[c*3+0];
  float ry = x[r*3+1] - x[c*3+1];
  float rz = x[r*3+2] - x[c*3+2];
  float* arow = agg + (size_t)r * HID;
  for (int ch = 0; ch < HID; ++ch) {
    float h = sb[ch] + sW[ch*3]*rx + sW[ch*3+1]*ry + sW[ch*3+2]*rz;
    atomicMaxF(arow + ch, h);
  }
}

__global__ void feat_kernel(const float* __restrict__ x,
                            const float* __restrict__ Wp,
                            const float* __restrict__ bp,
                            const float* __restrict__ agg,
                            float* __restrict__ feat) {
  int i = blockIdx.x * blockDim.x + threadIdx.x;
  if (i >= NV * HID) return;
  int v = i >> 6, ch = i & 63;
  float a = agg[i];
  if (a < -1e37f) a = 0.0f;
  feat[i] = bp[ch] + Wp[ch*3]*x[v*3] + Wp[ch*3+1]*x[v*3+1] + Wp[ch*3+2]*x[v*3+2] + a;
}

// --------------------------------------------- stage 3a: q/k projections via WMMA f16
// 8 waves / block; one wave per 16-row tile of feat.
__global__ void qk_wmma_kernel(const float* __restrict__ feat,
                               const float* __restrict__ Wq,
                               const float* __restrict__ Wk,
                               float* __restrict__ qb,
                               float* __restrict__ kb) {
  const int wave = threadIdx.x >> 5;
  const int lane = threadIdx.x & 31;
  const int hf   = lane >> 4;
  const int l16  = lane & 15;
  const int r0   = (blockIdx.x * 8 + wave) * 16;

  // A fragments: feat rows r0..r0+15, split K=0..31 / K=32..63
  v16h alo, ahi;
  const float* frow = feat + (size_t)(r0 + l16) * HID;
#pragma unroll
  for (int e = 0; e < 16; ++e) {
    int i = e >> 1, p = e & 1;
    int k = (i < 4) ? (8*hf + 2*i + p) : (16 + 8*hf + 2*(i-4) + p);
    alo[e] = (_Float16)frow[k];
    ahi[e] = (_Float16)frow[k + 32];
  }

  for (int nt = 0; nt < 4; ++nt) {
    const int n0 = nt * 16;
    // B[k][n] = W[(n0+n)*64 + k]; lane layout: K = e<8 ? 8*hf + e : 16 + 8*hf + (e-8) (+32 for hi), N = l16
    v16h bqlo, bqhi, bklo, bkhi;
    const float* wq = Wq + (size_t)(n0 + l16) * HID;
    const float* wk = Wk + (size_t)(n0 + l16) * HID;
#pragma unroll
    for (int e = 0; e < 16; ++e) {
      int k = (e < 8) ? (8*hf + e) : (16 + 8*hf + (e - 8));
      bqlo[e] = (_Float16)wq[k];
      bqhi[e] = (_Float16)wq[k + 32];
      bklo[e] = (_Float16)wk[k];
      bkhi[e] = (_Float16)wk[k + 32];
    }
    v8f cq = {};
    cq = __builtin_amdgcn_wmma_f32_16x16x32_f16(false, alo, false, bqlo, (short)0, cq, false, false);
    cq = __builtin_amdgcn_wmma_f32_16x16x32_f16(false, ahi, false, bqhi, (short)0, cq, false, false);
    v8f ck = {};
    ck = __builtin_amdgcn_wmma_f32_16x16x32_f16(false, alo, false, bklo, (short)0, ck, false, false);
    ck = __builtin_amdgcn_wmma_f32_16x16x32_f16(false, ahi, false, bkhi, (short)0, ck, false, false);
#pragma unroll
    for (int r = 0; r < 8; ++r) {
      int row = r0 + r + 8*hf;
      int col = n0 + l16;
      qb[(size_t)row * HID + col] = cq[r];
      kb[(size_t)row * HID + col] = ck[r];
    }
  }
}

// --------------------------------------------- stage 3b: edge attention + softmax
__global__ void att_dot_kernel(const float* __restrict__ qb, const float* __restrict__ kb,
                               const long long* __restrict__ erow, const long long* __restrict__ ecol,
                               float* __restrict__ att, float* __restrict__ mrow) {
  int wave = threadIdx.x >> 5, lane = threadIdx.x & 31;
  int e = blockIdx.x * 8 + wave;
  if (e >= NE) return;
  int r = (int)erow[e], c = (int)ecol[e];
  const float* qr = qb + (size_t)r * HID;
  const float* kc = kb + (size_t)c * HID;
  float s = qr[lane]*kc[lane] + qr[lane+32]*kc[lane+32];
  for (int off = 16; off > 0; off >>= 1) s += __shfl_xor(s, off, 32);
  if (lane == 0) { att[e] = s; atomicMaxF(&mrow[r], s); }
}

__global__ void att_exp_kernel(float* __restrict__ att, float* __restrict__ srow,
                               const float* __restrict__ mrow,
                               const long long* __restrict__ erow) {
  int e = blockIdx.x * blockDim.x + threadIdx.x;
  if (e >= NE) return;
  int r = (int)erow[e];
  float v = expf(att[e] - mrow[r]);
  att[e] = v;
  atomicAdd(&srow[r], v);
}

__global__ void att_norm_kernel(float* __restrict__ att, const float* __restrict__ srow,
                                const long long* __restrict__ erow) {
  int e = blockIdx.x * blockDim.x + threadIdx.x;
  if (e >= NE) return;
  att[e] /= srow[(int)erow[e]];
}

// --------------------------------------------- stage 4: sparse (S@A)@S^T
__global__ void deg_kernel(const long long* __restrict__ erow, const long long* __restrict__ ecol,
                           int* __restrict__ degA, int* __restrict__ degC) {
  int e = blockIdx.x * blockDim.x + threadIdx.x;
  if (e >= NE) return;
  atomicAdd(&degA[(int)erow[e]], 1);
  atomicAdd(&degC[(int)ecol[e]], 1);
}

__global__ void scan4096_kernel(const int* __restrict__ deg, int* __restrict__ ptr) {
  __shared__ int part[1024];
  int t = threadIdx.x;
  int b = t * 4;
  int d0 = deg[b], d1 = deg[b+1], d2 = deg[b+2], d3 = deg[b+3];
  part[t] = d0 + d1 + d2 + d3;
  __syncthreads();
  for (int off = 1; off < 1024; off <<= 1) {
    int v = part[t];
    int u = (t >= off) ? part[t - off] : 0;
    __syncthreads();
    part[t] = v + u;
    __syncthreads();
  }
  int excl = (t == 0) ? 0 : part[t - 1];
  ptr[b]   = excl;
  ptr[b+1] = excl + d0;
  ptr[b+2] = excl + d0 + d1;
  ptr[b+3] = excl + d0 + d1 + d2;
  if (t == 1023) ptr[NV] = part[1023];
}

__global__ void csr_fill_kernel(const long long* __restrict__ erow, const long long* __restrict__ ecol,
                                const float* __restrict__ att,
                                const int* __restrict__ ptrA, int* __restrict__ curA, int* __restrict__ colsA,
                                const int* __restrict__ ptrC, int* __restrict__ curC,
                                int* __restrict__ rowsS, float* __restrict__ valsS) {
  int e = blockIdx.x * blockDim.x + threadIdx.x;
  if (e >= NE) return;
  int r = (int)erow[e], c = (int)ecol[e];
  int pa = atomicAdd(&curA[r], 1);
  colsA[ptrA[r] + pa] = c;
  int pc = atomicAdd(&curC[c], 1);
  rowsS[ptrC[c] + pc] = r;
  valsS[ptrC[c] + pc] = att[e];
}

// out[i][j] += S[i][m] * A[m][k] * S[j][k]; one wave per S-edge (i,m).
__global__ void scatter_kernel(const long long* __restrict__ erow, const long long* __restrict__ ecol,
                               const float* __restrict__ att,
                               const int* __restrict__ ptrA, const int* __restrict__ colsA,
                               const int* __restrict__ ptrC, const int* __restrict__ rowsS,
                               const float* __restrict__ valsS,
                               float* __restrict__ out) {
  int wave = threadIdx.x >> 5, lane = threadIdx.x & 31;
  int e = blockIdx.x * 8 + wave;
  if (e >= NE) return;
  int i = (int)erow[e], m = (int)ecol[e];
  float v = att[e];
  float* orow = out + (size_t)i * NV;
  __builtin_prefetch(orow, 1, 0);        // global_prefetch_b8 hint on the hot output row
  int a0 = ptrA[m], a1 = ptrA[m+1];
  for (int idx = a0; idx < a1; ++idx) {
    int k = colsA[idx];
    int s0 = ptrC[k], s1 = ptrC[k+1];
    for (int j = s0 + lane; j < s1; j += 32) {
      atomicAdd(&orow[rowsS[j]], v * valsS[j]);
    }
  }
}

// ---------------------------------------------------------------- launcher
extern "C" void kernel_launch(void* const* d_in, const int* in_sizes, int n_in,
                              void* d_out, int out_size, void* d_ws, size_t ws_size,
                              hipStream_t stream) {
  (void)in_sizes; (void)n_in; (void)out_size; (void)ws_size;
  const float*     x    = (const float*)d_in[0];
  const long long* eidx = (const long long*)d_in[1];
  const long long* erow = eidx;
  const long long* ecol = eidx + NE;
  const float* Wt = (const float*)d_in[2];
  const float* bt = (const float*)d_in[3];
  const float* Wp = (const float*)d_in[4];
  const float* bp = (const float*)d_in[5];
  const float* Wq = (const float*)d_in[6];
  const float* Wk = (const float*)d_in[7];
  float* out = (float*)d_out;

  char* ws = (char*)d_ws;
  size_t o = 0;
  auto take = [&](size_t bytes) -> void* {
    void* p = ws + o;
    o += (bytes + 255) & ~(size_t)255;
    return p;
  };
  float* sq    = (float*)take(NV * sizeof(float));
  int*   knn   = (int*)  take((size_t)NV * KNN * sizeof(int));
  float* agg   = (float*)take((size_t)NV * HID * sizeof(float));
  float* feat  = (float*)take((size_t)NV * HID * sizeof(float));
  float* qb    = (float*)take((size_t)NV * HID * sizeof(float));
  float* kb    = (float*)take((size_t)NV * HID * sizeof(float));
  float* att   = (float*)take(NE * sizeof(float));
  float* mrow  = (float*)take(NV * sizeof(float));
  float* srow  = (float*)take(NV * sizeof(float));
  int*   degA  = (int*)  take(NV * sizeof(int));
  int*   ptrA  = (int*)  take((NV + 1) * sizeof(int));
  int*   curA  = (int*)  take(NV * sizeof(int));
  int*   degC  = (int*)  take(NV * sizeof(int));
  int*   ptrC  = (int*)  take((NV + 1) * sizeof(int));
  int*   curC  = (int*)  take(NV * sizeof(int));
  int*   colsA = (int*)  take(NE * sizeof(int));
  int*   rowsS = (int*)  take(NE * sizeof(int));
  float* valsS = (float*)take(NE * sizeof(float));

  // init
  sq_kernel<<<(NV + 255) / 256, 256, 0, stream>>>(x, sq);
  fill_f32_kernel<<<(NV * HID + 255) / 256, 256, 0, stream>>>(agg, -BIGF, NV * HID);
  fill_f32_kernel<<<(NV + 255) / 256, 256, 0, stream>>>(mrow, -BIGF, NV);
  fill_f32_kernel<<<(NV + 255) / 256, 256, 0, stream>>>(srow, 0.0f, NV);
  fill_i32_kernel<<<(NV + 255) / 256, 256, 0, stream>>>(degA, 0, NV);
  fill_i32_kernel<<<(NV + 255) / 256, 256, 0, stream>>>(curA, 0, NV);
  fill_i32_kernel<<<(NV + 255) / 256, 256, 0, stream>>>(degC, 0, NV);
  fill_i32_kernel<<<(NV + 255) / 256, 256, 0, stream>>>(curC, 0, NV);

  // stage 1: kNN (D2 staged through d_out, dead until final stage)
  dist_wmma_kernel<<<NV / 16, 256, 0, stream>>>(x, sq, out);
  knn_select_kernel<<<NV, 256, 0, stream>>>(out, knn);

  // stage 2: DevConv
  devconv_kernel<<<(NEE + 255) / 256, 256, 0, stream>>>(x, erow, ecol, knn, Wt, bt, agg);
  feat_kernel<<<(NV * HID + 255) / 256, 256, 0, stream>>>(x, Wp, bp, agg, feat);

  // stage 3: attention
  qk_wmma_kernel<<<NV / 128, 256, 0, stream>>>(feat, Wq, Wk, qb, kb);
  att_dot_kernel<<<(NE + 7) / 8, 256, 0, stream>>>(qb, kb, erow, ecol, att, mrow);
  att_exp_kernel<<<(NE + 255) / 256, 256, 0, stream>>>(att, srow, mrow, erow);
  att_norm_kernel<<<(NE + 255) / 256, 256, 0, stream>>>(att, srow, erow);

  // stage 4: sparse (S@A)@S^T
  deg_kernel<<<(NE + 255) / 256, 256, 0, stream>>>(erow, ecol, degA, degC);
  scan4096_kernel<<<1, 1024, 0, stream>>>(degA, ptrA);
  scan4096_kernel<<<1, 1024, 0, stream>>>(degC, ptrC);
  csr_fill_kernel<<<(NE + 255) / 256, 256, 0, stream>>>(erow, ecol, att, ptrA, curA, colsA,
                                                        ptrC, curC, rowsS, valsS);
  zero4_kernel<<<(NV * NV / 4 + 255) / 256, 256, 0, stream>>>((float4*)out, NV * NV / 4);
  scatter_kernel<<<(NE + 7) / 8, 256, 0, stream>>>(erow, ecol, att, ptrA, colsA,
                                                   ptrC, rowsS, valsS, out);
}
